// Baseline_60516089201168
// MI455X (gfx1250) — matrix operand order repaired and checked by reading, not compiled, not run on hardware
//
#include <hip/hip_runtime.h>

// ---------------------------------------------------------------------------
// Model dims
#define BB 32
#define TT 2048
#define HH 200
#define CC 256
#define N4H 800          // 4*H
#define MTOT (BB*TT)     // 65536 rows

typedef __attribute__((ext_vector_type(16))) __bf16 v16bf;
typedef __attribute__((ext_vector_type(8)))  __bf16 v8bf;
typedef __attribute__((ext_vector_type(8)))  float  v8f;
typedef __attribute__((ext_vector_type(4)))  unsigned int v4u;
typedef __attribute__((ext_vector_type(8)))  int v8i;
typedef __attribute__((ext_vector_type(4)))  int v4i;

static __device__ __forceinline__ v16bf bfcat(v8bf lo, v8bf hi) {
    return __builtin_shufflevector(lo, hi, 0,1,2,3,4,5,6,7,8,9,10,11,12,13,14,15);
}
static __device__ __forceinline__ v8f wmma_bf16(v16bf a, v16bf b, v8f c) {
    return __builtin_amdgcn_wmma_f32_16x16x32_bf16(false, a, false, b, (short)0, c, false, false);
}
static __device__ __forceinline__ float sigf(float x) {
    return 1.0f / (1.0f + __expf(-x));
}

// ---------------------------------------------------------------------------
// Kernel 0: weight conversion + fragment swizzle.
// Fragment layout: element e of lane L of k-chunk kc of n-tile nt holds
//   W[K = kc*32 + 8*(L>>4) + (e&7) + 16*(e>>3)][n = nt*16 + (L&15)]   (bf16)
// LSTM recurrent weights are emitted as:
//   usl: contiguous LDS image of k-chunks 0..5  (((nt*6+kc)*32+L)*16) —
//        TDM-copied verbatim into LDS by lstm_rec
//   u6 : k-chunk 6 fragments ((nt*32+L)*16), K=192..223 zero-padded past 199
// Gate-interleaved column permutation: n -> gate=n&3, u=n>>2, src=gate*200+u
// (i,j,f,o order; forget-gate +1.0 folded into bias).
__global__ __launch_bounds__(256) void prep_weights(
    const float* k2w, const float* k3w,
    const float* Wf, const float* bfv, const float* Wb, const float* bbv,
    __bf16* k2s, __bf16* k3s, __bf16* wxsf, __bf16* wxsb,
    __bf16* uslf, __bf16* uslb, __bf16* u6f, __bf16* u6b,
    float* bif, float* bib)
{
    const int SK2  = 16*24*512;  // 196608
    const int SK3  = 16*8*512;   // 65536
    const int SWX  = 50*8*512;   // 204800
    const int SUSL = 50*6*512;   // 153600  (K chunks 0..5, K < 192)
    const int SU6  = 50*512;     // 25600   (K chunk 6, K = 192..223)
    const int total = SK2 + SK3 + 2*SWX + 2*SUSL + 2*SU6 + 2*N4H;
    for (int i = blockIdx.x*blockDim.x + threadIdx.x; i < total;
         i += gridDim.x*blockDim.x) {
        int idx = i;
        if (idx < SK2) {
            int e = idx & 15, L = (idx >> 4) & 31;
            int kc = (idx >> 9) % 24, nt = idx / (512*24);
            int K = kc*32 + 8*(L >> 4) + (e & 7) + 16*(e >> 3), n = (nt << 4) + (L & 15);
            k2s[idx] = (__bf16)k2w[K*CC + n];
            continue;
        }
        idx -= SK2;
        if (idx < SK3) {
            int e = idx & 15, L = (idx >> 4) & 31;
            int kc = (idx >> 9) & 7, nt = idx / (512*8);
            int K = kc*32 + 8*(L >> 4) + (e & 7) + 16*(e >> 3), n = (nt << 4) + (L & 15);
            k3s[idx] = (__bf16)k3w[K*CC + n];
            continue;
        }
        idx -= SK3;
        if (idx < 2*SWX) {
            int d = idx / SWX, s2 = idx % SWX;
            int e = s2 & 15, L = (s2 >> 4) & 31;
            int kc = (s2 >> 9) & 7, nt = s2 / (512*8);
            int K = kc*32 + 8*(L >> 4) + (e & 7) + 16*(e >> 3), n = (nt << 4) + (L & 15);
            int src = (n & 3)*HH + (n >> 2);
            const float* W = d ? Wb : Wf;
            (d ? wxsb : wxsf)[s2] = (__bf16)W[K*N4H + src];
            continue;
        }
        idx -= 2*SWX;
        if (idx < 2*SUSL) {
            int d = idx / SUSL, s2 = idx % SUSL;
            int e = s2 & 15, L = (s2 >> 4) & 31;
            int kc = (s2 >> 9) % 6, nt = s2 / (512*6);
            int K = kc*32 + 8*(L >> 4) + (e & 7) + 16*(e >> 3), n = (nt << 4) + (L & 15);
            int src = (n & 3)*HH + (n >> 2);
            const float* W = d ? Wb : Wf;
            (d ? uslb : uslf)[s2] = (__bf16)W[(CC + K)*N4H + src];
            continue;
        }
        idx -= 2*SUSL;
        if (idx < 2*SU6) {
            int d = idx / SU6, s2 = idx % SU6;
            int e = s2 & 15, L = (s2 >> 4) & 31, nt = s2 >> 9;
            int K = 192 + 8*(L >> 4) + (e & 7) + 16*(e >> 3), n = (nt << 4) + (L & 15);
            int src = (n & 3)*HH + (n >> 2);
            const float* W = d ? Wb : Wf;
            float v = (K < HH) ? W[(CC + K)*N4H + src] : 0.0f;
            (d ? u6b : u6f)[s2] = (__bf16)v;
            continue;
        }
        idx -= 2*SU6;
        {
            int d = idx / N4H, n = idx % N4H;
            int gate = n & 3, src = gate*HH + (n >> 2);
            const float* bsrc = d ? bbv : bfv;
            (d ? bib : bif)[n] = bsrc[src] + (gate == 2 ? 1.0f : 0.0f);
        }
    }
}

// ---------------------------------------------------------------------------
// Kernel 1: 1->C pointwise convs (conv1 relu -> bf16, conv1a relu -> f32 skip)
__global__ __launch_bounds__(256) void conv1_pw(
    const float* sig, const float* k1w, const float* k1aw, const float* k1ab,
    __bf16* c1, float* c1a)
{
    int m = blockIdx.x;       // (b*T + t)
    int c = threadIdx.x;
    float x = sig[m];
    float a = x * k1w[c];           a = a > 0.f ? a : 0.f;
    float v = x * k1aw[c] + k1ab[c]; v = v > 0.f ? v : 0.f;
    c1 [(size_t)m*CC + c] = (__bf16)a;
    c1a[(size_t)m*CC + c] = v;
}

// ---------------------------------------------------------------------------
// Kernel 2: conv2 (3-tap, SAME) as a K=768 GEMM over shifted rows of c1.
// WG = 256 thr (8 waves); tile M=32 x N=256. wave -> (mt = w/4, 4 n-tiles).
__global__ __launch_bounds__(256) void conv2_gemm(
    const __bf16* c1, const __bf16* k2s, __bf16* out)
{
    int tid = threadIdx.x, wave = tid >> 5, L = tid & 31, g = L >> 4, ln = L & 15;
    int mt = wave >> 2, ntBase = (wave & 3) * 4;
    int m0 = blockIdx.x*32 + mt*16;
    int b = m0 / TT, t = (m0 % TT) + ln;      // A-row for this lane is M=ln
    v8f zero = {};
    v8f acc[4]; acc[0]=zero; acc[1]=zero; acc[2]=zero; acc[3]=zero;
    for (int kc = 0; kc < 24; ++kc) {
        int kA = kc*32 + 8*g;
        int w  = kA >> 8;            // tap index 0..2 (runs never cross taps)
        int c0 = kA & 255;
        int t2 = t + w - 1;
        v8bf lo = {}, hi = {};
        if (t2 >= 0 && t2 < TT) {
            const __bf16* p = c1 + ((size_t)b*TT + t2)*CC + c0;
            lo = *(const v8bf*)p; hi = *(const v8bf*)(p + 16);
        }
        v16bf a = bfcat(lo, hi);
        #pragma unroll
        for (int j = 0; j < 4; ++j) {
            int nt = ntBase + j;
            const v8bf* bp = (const v8bf*)(k2s + ((size_t)(nt*24 + kc)*32 + L)*16);
            acc[j] = wmma_bf16(a, bfcat(bp[0], bp[1]), acc[j]);
        }
    }
    #pragma unroll
    for (int j = 0; j < 4; ++j) {
        int nt = ntBase + j;
        #pragma unroll
        for (int v = 0; v < 8; ++v) {
            int row = m0 + v + 8*g;
            float x = acc[j][v]; x = x > 0.f ? x : 0.f;
            out[(size_t)row*CC + nt*16 + ln] = (__bf16)x;
        }
    }
}

// ---------------------------------------------------------------------------
// Kernel 3: conv3 (1x1 GEMM, K=256) + c1a skip + relu -> enc (bf16)
__global__ __launch_bounds__(256) void conv3_gemm(
    const __bf16* cv2, const __bf16* k3s, const float* c1a, __bf16* enc)
{
    int tid = threadIdx.x, wave = tid >> 5, L = tid & 31, g = L >> 4, ln = L & 15;
    int mt = wave >> 2, ntBase = (wave & 3) * 4;
    int m0 = blockIdx.x*32 + mt*16;
    v8f zero = {};
    v8f acc[4]; acc[0]=zero; acc[1]=zero; acc[2]=zero; acc[3]=zero;
    for (int kc = 0; kc < 8; ++kc) {
        const __bf16* p = cv2 + ((size_t)(m0 + ln))*CC + kc*32 + 8*g;
        v16bf a = bfcat(*(const v8bf*)p, *(const v8bf*)(p + 16));
        #pragma unroll
        for (int j = 0; j < 4; ++j) {
            int nt = ntBase + j;
            const v8bf* bp = (const v8bf*)(k3s + ((size_t)(nt*8 + kc)*32 + L)*16);
            acc[j] = wmma_bf16(a, bfcat(bp[0], bp[1]), acc[j]);
        }
    }
    #pragma unroll
    for (int j = 0; j < 4; ++j) {
        int nt = ntBase + j;
        #pragma unroll
        for (int v = 0; v < 8; ++v) {
            size_t row = (size_t)(m0 + v + 8*g);
            float x = acc[j][v] + c1a[row*CC + nt*16 + ln];
            x = x > 0.f ? x : 0.f;
            enc[row*CC + nt*16 + ln] = (__bf16)x;
        }
    }
}

// ---------------------------------------------------------------------------
// Kernel 4: LSTM input projection  xhat = enc @ Wx + b  (gate-interleaved cols)
// WG = 320 thr (10 waves); tile M=32 x N=800; wave -> (mt = w&1, 10 n-tiles)
__global__ __launch_bounds__(320) void xhat_gemm(
    const __bf16* enc, const __bf16* wxsf, const __bf16* wxsb,
    const float* bif, const float* bib, float* xhf, float* xhb)
{
    int dir = blockIdx.y;
    const __bf16* wxs = dir ? wxsb : wxsf;
    const float*  bi  = dir ? bib  : bif;
    float*        xh  = dir ? xhb  : xhf;
    int tid = threadIdx.x, wave = tid >> 5, L = tid & 31, g = L >> 4, ln = L & 15;
    int mt = wave & 1, ntBase = (wave >> 1) * 10;
    int m0 = blockIdx.x*32 + mt*16;
    v8f zero = {};
    v8f acc[10];
    #pragma unroll
    for (int j = 0; j < 10; ++j) acc[j] = zero;
    for (int kc = 0; kc < 8; ++kc) {
        const __bf16* p = enc + ((size_t)(m0 + ln))*CC + kc*32 + 8*g;
        v16bf a = bfcat(*(const v8bf*)p, *(const v8bf*)(p + 16));
        #pragma unroll
        for (int j = 0; j < 10; ++j) {
            int nt = ntBase + j;
            const v8bf* bp = (const v8bf*)(wxs + ((size_t)(nt*8 + kc)*32 + L)*16);
            acc[j] = wmma_bf16(a, bfcat(bp[0], bp[1]), acc[j]);
        }
    }
    #pragma unroll
    for (int j = 0; j < 10; ++j) {
        int nt = ntBase + j;
        float bcol = bi[nt*16 + ln];
        #pragma unroll
        for (int v = 0; v < 8; ++v) {
            size_t row = (size_t)(m0 + v + 8*g);
            xh[row*N4H + nt*16 + ln] = acc[j][v] + bcol;
        }
    }
}

// ---------------------------------------------------------------------------
// Kernel 5: persistent recurrence.  4 WGs = {fwd,bwd} x {batch rows 0-15,16-31}.
// U k-chunks 0..5 (307,200 B fragment image) staged into LDS with one
// TENSOR_LOAD_TO_LDS (TDM, 2-D descriptor: 75 rows x 4096 B, 8-byte elems),
// chunk 6 held in registers.  Per step: z = h @ U (bf16 WMMA) + xhat; gate
// exchange via ds_swizzle quad-broadcast; h exchanged through LDS.
#define LDS_U_BYTES (50*6*512*2)   // 307200
#define LDS_H_OFF   LDS_U_BYTES
#define LDS_TOTAL   (LDS_U_BYTES + 16*224*2)   // + 7168 = 314368
__global__ __launch_bounds__(320) void lstm_rec(
    const float* xhatf, const float* xhatb,
    const __bf16* uslf, const __bf16* uslb,
    const __bf16* u6f, const __bf16* u6b,
    const int* lens, __bf16* hsf, __bf16* hsb)
{
    extern __shared__ char smem[];
    __bf16* uLDS = (__bf16*)smem;
    __bf16* hLDS = (__bf16*)(smem + LDS_H_OFF);

    int dir = blockIdx.x >> 1;
    int b0  = (blockIdx.x & 1) * 16;
    const float*  xhat = dir ? xhatb : xhatf;
    const __bf16* usl  = dir ? uslb  : uslf;
    const __bf16* u6   = dir ? u6b   : u6f;
    __bf16*       hs   = dir ? hsb   : hsf;

    int tid = threadIdx.x, wave = tid >> 5, L = tid & 31, g = L >> 4, ln = L & 15;

    // ---- TDM: stage the U LDS-image (kc 0..5) into LDS, issued by wave 0 ----
    if (wave == 0) {
        unsigned long long ga = (unsigned long long)(size_t)usl;
        // D# group 0: count=1, lds_addr=0 (dynamic LDS base), global addr,
        // type=2 ("image") in bits [127:126]
        v4u g0 = {};
        g0[0] = 1u;                                   // count=1, is_restore=0
        g0[1] = 0u;                                   // lds_addr
        g0[2] = (unsigned)(ga & 0xFFFFFFFFu);         // global_addr[31:0]
        g0[3] = (unsigned)((ga >> 32) & 0x1FFFFFFu) | 0x80000000u; // type=2
        // D# group 1: data_size=8B; tensor_dim0=512 units (4096 B),
        // tensor_dim1=75, tile_dim0=512, tile_dim1=75, dim0_stride=512
        v8i g1 = {};
        g1[0] = 3 << 16;                 // data_size = 8 bytes
        g1[1] = 512 << 16;               // tensor_dim0[15:0] in [31:16]
        g1[2] = 75 << 16;                // tensor_dim1[15:0] in [31:16]
        g1[3] = 512 << 16;               // tile_dim0 in [31:16]
        g1[4] = 75;                      // tile_dim1
        g1[5] = 512;                     // tensor_dim0_stride[31:0]
        g1[6] = 512 << 16;               // tensor_dim1_stride[15:0]
        g1[7] = 0;
        v4i g2 = {};
        v4i g3 = {};
#if defined(__clang_major__) && __clang_major__ >= 23
        v8i g4 = {};
        __builtin_amdgcn_tensor_load_to_lds(g0, g1, g2, g3, g4, 0);
#else
        __builtin_amdgcn_tensor_load_to_lds(g0, g1, g2, g3, 0);
#endif
        __builtin_amdgcn_s_wait_tensorcnt(0);
    }
    // zero h (16 rows x 224 padded units)
    int4 zz = make_int4(0, 0, 0, 0);
    for (int i = tid; i < 448; i += 320) ((int4*)hLDS)[i] = zz;

    // k-chunk 6 B-fragments held in registers (step-invariant)
    v16bf b6[5];
    #pragma unroll
    for (int j = 0; j < 5; ++j) {
        int nt = wave*5 + j;
        const v8bf* p = (const v8bf*)(u6 + ((size_t)(nt*32 + L))*16);
        b6[j] = bfcat(p[0], p[1]);
    }
    int lenr[8];
    #pragma unroll
    for (int v = 0; v < 8; ++v) lenr[v] = lens[b0 + v + 8*g];

    float cReg[5][8];
    #pragma unroll
    for (int j = 0; j < 5; ++j)
        #pragma unroll
        for (int v = 0; v < 8; ++v) cReg[j][v] = 0.f;

    __syncthreads();

    for (int s = 0; s < TT; ++s) {
        // A fragments from h (lane = row M, K runs at 8g and 16+8g)
        v16bf afr[7];
        #pragma unroll
        for (int kc = 0; kc < 7; ++kc) {
            const __bf16* hp = hLDS + ln*224 + kc*32 + 8*g;
            afr[kc] = bfcat(*(const v8bf*)hp, *(const v8bf*)(hp + 16));
        }
        __syncthreads();   // h reads done before anyone writes h_new

        #pragma unroll
        for (int j = 0; j < 5; ++j) {
            int nt = wave*5 + j;
            v8f acc = {};
            #pragma unroll
            for (int kc = 0; kc < 6; ++kc) {
                const v8bf* bp =
                    (const v8bf*)(uLDS + ((size_t)(nt*6 + kc)*32 + L)*16);
                acc = wmma_bf16(afr[kc], bfcat(bp[0], bp[1]), acc);
            }
            acc = wmma_bf16(afr[6], b6[j], acc);

            int q = ln >> 2;            // unit-in-tile
            int unit = nt*4 + q;
            #pragma unroll
            for (int v = 0; v < 8; ++v) {
                int r = v + 8*g, b = b0 + r, Lr = lenr[v];
                bool m = s < Lr;
                int p = (dir == 0) ? s : (m ? (Lr - 1 - s) : s);
                float z = acc[v] + xhat[((size_t)b*TT + p)*N4H + nt*16 + ln];
                // quad-broadcast the 4 gate columns (i,j,f,o) via ds_swizzle
                float zi = __int_as_float(
                    __builtin_amdgcn_ds_swizzle(__float_as_int(z), 0x8000));
                float zj = __int_as_float(
                    __builtin_amdgcn_ds_swizzle(__float_as_int(z), 0x8055));
                float zf = __int_as_float(
                    __builtin_amdgcn_ds_swizzle(__float_as_int(z), 0x80AA));
                float zo = __int_as_float(
                    __builtin_amdgcn_ds_swizzle(__float_as_int(z), 0x80FF));
                float ig = sigf(zi), jg = tanhf(zj);
                float fg = sigf(zf), og = sigf(zo);   // +1.0 folded into bias
                float cn = cReg[j][v]*fg + ig*jg;
                float hn = tanhf(cn)*og;
                if (m) cReg[j][v] = cn;
                if ((ln & 3) == 0) {
                    if (m) hLDS[r*224 + unit] = (__bf16)hn;
                    hs[((size_t)b*TT + p)*HH + unit] = (__bf16)(m ? hn : 0.f);
                }
            }
            if (s + 1 < TT)
                __builtin_prefetch(
                    xhat + ((size_t)(b0 + 8*g)*TT + (s + 1))*N4H + nt*16 + ln,
                    0, 0);
        }
        __syncthreads();   // h_new visible before next step's reads
    }
}

// ---------------------------------------------------------------------------
// Kernel 6: logits = [fw, bw] @ Wd + bd  (K=400, N=5; bandwidth-bound scalar)
__global__ __launch_bounds__(256) void logits_k(
    const __bf16* hsf, const __bf16* hsb, const float* Wd, const float* bd,
    float* out)
{
    int m = blockIdx.x*256 + threadIdx.x;
    if (m >= MTOT) return;
    float a[5];
    #pragma unroll
    for (int k = 0; k < 5; ++k) a[k] = bd[k];
    const __bf16* pf = hsf + (size_t)m*HH;
    const __bf16* pb = hsb + (size_t)m*HH;
    for (int u = 0; u < HH; ++u) {
        float hf = (float)pf[u];
        float hb = (float)pb[u];
        #pragma unroll
        for (int k = 0; k < 5; ++k) {
            a[k] += hf * Wd[u*5 + k];
            a[k] += hb * Wd[(HH + u)*5 + k];
        }
    }
    #pragma unroll
    for (int k = 0; k < 5; ++k) out[(size_t)m*5 + k] = a[k];
}

// ---------------------------------------------------------------------------
extern "C" void kernel_launch(void* const* d_in, const int* in_sizes, int n_in,
                              void* d_out, int out_size, void* d_ws,
                              size_t ws_size, hipStream_t stream)
{
    const float* signals = (const float*)d_in[0];
    const int*   lens    = (const int*)  d_in[1];
    const float* k1w     = (const float*)d_in[2];
    const float* k1aw    = (const float*)d_in[3];
    const float* k1ab    = (const float*)d_in[4];
    const float* k2w     = (const float*)d_in[5];
    const float* k3w     = (const float*)d_in[6];
    const float* Wf      = (const float*)d_in[7];
    const float* bfv     = (const float*)d_in[8];
    const float* Wb      = (const float*)d_in[9];
    const float* bbv     = (const float*)d_in[10];
    const float* Wd      = (const float*)d_in[11];
    const float* bd      = (const float*)d_in[12];
    float* out = (float*)d_out;

    char* ws = (char*)d_ws;
    size_t off = 0;
    auto carve = [&](size_t bytes) -> char* {
        char* p = ws + off;
        off += (bytes + 511) & ~(size_t)511;
        return p;
    };
    __bf16* c1   = (__bf16*)carve((size_t)MTOT*CC*2);
    float*  c1a  = (float*) carve((size_t)MTOT*CC*4);
    __bf16* cv2  = (__bf16*)carve((size_t)MTOT*CC*2);
    __bf16* enc  = (__bf16*)carve((size_t)MTOT*CC*2);
    __bf16* k2s  = (__bf16*)carve((size_t)16*24*512*2);
    __bf16* k3s  = (__bf16*)carve((size_t)16*8*512*2);
    __bf16* wxsf = (__bf16*)carve((size_t)50*8*512*2);
    __bf16* wxsb = (__bf16*)carve((size_t)50*8*512*2);
    __bf16* uslf = (__bf16*)carve((size_t)50*6*512*2);
    __bf16* uslb = (__bf16*)carve((size_t)50*6*512*2);
    __bf16* u6f  = (__bf16*)carve((size_t)50*512*2);
    __bf16* u6b  = (__bf16*)carve((size_t)50*512*2);
    float*  bif  = (float*) carve((size_t)N4H*4);
    float*  bib  = (float*) carve((size_t)N4H*4);
    float*  xhf  = (float*) carve((size_t)MTOT*N4H*4);
    float*  xhb  = (float*) carve((size_t)MTOT*N4H*4);
    __bf16* hsf  = (__bf16*)carve((size_t)MTOT*HH*2);
    __bf16* hsb  = (__bf16*)carve((size_t)MTOT*HH*2);

    prep_weights<<<1024, 256, 0, stream>>>(k2w, k3w, Wf, bfv, Wb, bbv,
                                           k2s, k3s, wxsf, wxsb,
                                           uslf, uslb, u6f, u6b, bif, bib);
    conv1_pw<<<MTOT, 256, 0, stream>>>(signals, k1w, k1aw, k1ab, c1, c1a);
    conv2_gemm<<<MTOT/32, 256, 0, stream>>>(c1, k2s, cv2);
    conv3_gemm<<<MTOT/32, 256, 0, stream>>>(cv2, k3s, c1a, enc);
    xhat_gemm<<<dim3(MTOT/32, 2), 320, 0, stream>>>(enc, wxsf, wxsb,
                                                    bif, bib, xhf, xhb);
    lstm_rec<<<4, 320, LDS_TOTAL, stream>>>(xhf, xhb, uslf, uslb, u6f, u6b,
                                            lens, hsf, hsb);
    logits_k<<<MTOT/256, 256, 0, stream>>>(hsf, hsb, Wd, bd, out);
    (void)in_sizes; (void)n_in; (void)out_size; (void)ws_size;
}
